// GLAT_42236708389036
// MI455X (gfx1250) — matrix operand order repaired and checked by reading, not compiled, not run on hardware
//
#include <hip/hip_runtime.h>
#include <stdint.h>

// Problem constants (match reference)
#define BDIM   64
#define NSEQ   256
#define DMODEL 1024
#define NHEAD  8
#define DKH    64
#define DINNER 2048
#define BN     (BDIM * NSEQ)   // 16384 rows
#define HD     (NHEAD * DKH)   // 512

static constexpr float kScale = 0.125f;   // 1/sqrt(64)
static constexpr float kNeg   = -1e9f;

typedef __attribute__((ext_vector_type(16))) __bf16 bf16x16;
typedef __attribute__((ext_vector_type(8)))  float  f32x8;

union FragB16 {
    bf16x16 v;
    uint4 q[2];
    unsigned short s[16];
};

__device__ __forceinline__ unsigned short f2b(float f) {
    union { float f; unsigned u; } x; x.f = f;
    unsigned r = x.u + 0x7FFFu + ((x.u >> 16) & 1u);   // round-to-nearest-even
    return (unsigned short)(r >> 16);
}

__device__ __forceinline__ f32x8 zero8() {
    f32x8 z = {0.f,0.f,0.f,0.f,0.f,0.f,0.f,0.f};
    return z;
}

// low 32 bits of a generic pointer into LDS == LDS byte offset (aperture layout)
__device__ __forceinline__ unsigned lds_off(const void* p) {
    return (unsigned)(unsigned long long)p;
}

// ---------------- elementwise f32 -> bf16 ----------------
__global__ void cvt_kernel(const float* __restrict__ in,
                           unsigned short* __restrict__ out, int n) {
    int i = blockIdx.x * blockDim.x + threadIdx.x;
    if (i < n) out[i] = f2b(in[i]);
}

// ---------------- transpose+convert: W[K,N] f32 -> Wt[N,K] bf16 ----------------
__global__ void tcvt_kernel(const float* __restrict__ in,
                            unsigned short* __restrict__ out, int K, int N) {
    int i = blockIdx.x * blockDim.x + threadIdx.x;
    if (i < K * N) {
        int k = i / N, n = i - k * N;
        out[(size_t)n * K + k] = f2b(in[i]);
    }
}

// ---------------- tiled bf16 GEMM: C[M,N] = A[M,K] * Bt[N,K]^T (+bias)(relu) ----
// 256 threads = 8 waves; 128x128 block tile; wave patch 32x64 (2x4 WMMA tiles).
// K staged 64-deep per LDS buffer: 16 WMMAs per barrier. Global->LDS staging via
// GLOBAL_LOAD_ASYNC_TO_LDS_B128, double buffered: fetch of stage i+1 overlaps
// the 16 WMMAs of stage i; one s_wait_asynccnt + one barrier per stage.
#define KSTEP 64
#define LDSTR 72   // 64 + 8 pad, byte stride 144 (16B multiple)
__global__ __launch_bounds__(256) void gemm_bf16_kernel(
    const unsigned short* __restrict__ A, int lda,
    const unsigned short* __restrict__ Bt, int ldb,
    const float* __restrict__ bias,
    float* __restrict__ Cf,
    unsigned short* __restrict__ Cb, int ldc, int colOff,
    int K, int doRelu)
{
    __shared__ __align__(16) unsigned short As[2][128 * LDSTR];
    __shared__ __align__(16) unsigned short Bs[2][128 * LDSTR];

    const int tid  = threadIdx.x;
    const int m0   = blockIdx.y * 128;
    const int n0   = blockIdx.x * 128;
    const int wid  = tid >> 5;
    const int lane = tid & 31;
    const int r    = lane & 15;
    const int h    = lane >> 4;
    const int wm   = (wid >> 1) * 32;   // 4 wave-rows of 32
    const int wn   = (wid & 1) * 64;    // 2 wave-cols of 64

    // per-thread staging assignment: 4 chunks of 8 bf16 per matrix per stage
    int rrow[4], rcol[4];
    #pragma unroll
    for (int i = 0; i < 4; ++i) {
        int c   = tid + i * 256;   // 0..1023
        rrow[i] = c >> 3;          // 0..127
        rcol[i] = (c & 7) * 8;     // 0..56
    }

    f32x8 acc[2][4];
    #pragma unroll
    for (int i = 0; i < 2; ++i)
        #pragma unroll
        for (int j = 0; j < 4; ++j) acc[i][j] = zero8();

    const int nk = K / KSTEP;

    // issue async loads of one 128x64 K-stage (k0) into buffer buf
    auto issue = [&](int k0, int buf) {
        #pragma unroll
        for (int i = 0; i < 4; ++i) {
            const unsigned short* ga =
                A + (size_t)(m0 + rrow[i]) * lda + k0 + rcol[i];
            unsigned la = lds_off(&As[buf][rrow[i] * LDSTR + rcol[i]]);
            asm volatile("global_load_async_to_lds_b128 %0, %1, off"
                         :: "v"(la), "v"(ga) : "memory");
            const unsigned short* gb =
                Bt + (size_t)(n0 + rrow[i]) * ldb + k0 + rcol[i];
            unsigned lb = lds_off(&Bs[buf][rrow[i] * LDSTR + rcol[i]]);
            asm volatile("global_load_async_to_lds_b128 %0, %1, off"
                         :: "v"(lb), "v"(gb) : "memory");
        }
    };

    issue(0, 0);

    for (int it = 0; it < nk; ++it) {
        const int cur = it & 1;
        asm volatile("s_wait_asynccnt 0" ::: "memory");  // my stage-it writes done
        __syncthreads();   // all waves: writes landed; previous-stage reads done
        if (it + 1 < nk) issue((it + 1) * KSTEP, cur ^ 1);

        #pragma unroll
        for (int kk = 0; kk < 2; ++kk) {     // two 32-deep WMMA sub-steps
            const int ko = kk * 32;
            FragB16 a[2], b[4];
            #pragma unroll
            for (int ti = 0; ti < 2; ++ti) { // A-frag: contiguous 16 at h*16
                const unsigned short* p =
                    &As[cur][(wm + ti * 16 + r) * LDSTR + ko];
                a[ti].q[0] = *(const uint4*)(p + h * 16);
                a[ti].q[1] = *(const uint4*)(p + h * 16 + 8);
            }
            #pragma unroll
            for (int tj = 0; tj < 4; ++tj) { // B-frag: contiguous 16 at h*16
                const unsigned short* p =
                    &Bs[cur][(wn + tj * 16 + r) * LDSTR + ko + h * 16];
                b[tj].q[0] = *(const uint4*)(p);
                b[tj].q[1] = *(const uint4*)(p + 8);
            }
            #pragma unroll
            for (int ti = 0; ti < 2; ++ti)
                #pragma unroll
                for (int tj = 0; tj < 4; ++tj)
                    acc[ti][tj] = __builtin_amdgcn_wmma_f32_16x16x32_bf16(
                        false, a[ti].v, false, b[tj].v, (short)0, acc[ti][tj],
                        false, false);
        }
    }

    #pragma unroll
    for (int ti = 0; ti < 2; ++ti)
        #pragma unroll
        for (int tj = 0; tj < 4; ++tj) {
            int col  = n0 + wn + tj * 16 + r;
            float bv = bias ? bias[col] : 0.f;
            #pragma unroll
            for (int vg = 0; vg < 8; ++vg) {
                int row = m0 + wm + ti * 16 + vg + h * 8;
                float v = acc[ti][tj][vg] + bv;
                if (doRelu) v = v > 0.f ? v : 0.f;
                if (Cf) Cf[(size_t)row * ldc + col] = v;
                if (Cb) Cb[(size_t)row * ldc + colOff + col] = f2b(v);
            }
        }
}

// ---------------- fused masked attention for one (b, h, 64-row q tile) --------
// grid = B*H*4, block = 128 (4 waves, each owns 16 query rows)
__global__ __launch_bounds__(128) void attn_kernel(
    const unsigned short* __restrict__ Q,
    const unsigned short* __restrict__ Kb,
    const unsigned short* __restrict__ Vb,
    const int* __restrict__ slf,
    const int* __restrict__ adj,
    int useAdj,
    unsigned short* __restrict__ Ctx)
{
    __shared__ __align__(16) float          Ssm[64 * 264];   // scores / probs (f32)
    __shared__ __align__(16) unsigned short Vt[64 * 264];    // V transposed [dk][key]

    const int tid  = threadIdx.x;
    const int blk  = blockIdx.x;
    const int qt   = blk & 3;
    const int hh   = (blk >> 2) & (NHEAD - 1);
    const int bb   = blk >> 5;
    const int q0   = qt * 64;
    const int wid  = tid >> 5;
    const int lane = tid & 31;
    const int r    = lane & 15;
    const int h    = lane >> 4;
    const size_t rowBase = (size_t)bb * NSEQ;
    const int colH = hh * DKH;

    // V -> LDS transposed
    for (int i = tid; i < NSEQ * DKH; i += 128) {
        int key = i >> 6, dk = i & 63;
        Vt[dk * 264 + key] = Vb[(rowBase + key) * HD + colH + dk];
    }

    // Q fragments for this wave's 16 rows (dk chunks kk=0,32)
    FragB16 aq[2];
    {
        const unsigned short* qp = &Q[(rowBase + q0 + wid * 16 + r) * HD + colH];
        #pragma unroll
        for (int kk = 0; kk < 2; ++kk) {
            aq[kk].q[0] = *(const uint4*)(qp + kk * 32 + h * 16);
            aq[kk].q[1] = *(const uint4*)(qp + kk * 32 + h * 16 + 8);
        }
    }

    // S = scale * Q K^T with mask, into LDS
    for (int nt = 0; nt < 16; ++nt) {
        f32x8 acc = zero8();
        #pragma unroll
        for (int kk = 0; kk < 2; ++kk) {
            FragB16 bk;   // B-frag lane = key row nt*16+r, contiguous dk at h*16
            const unsigned short* kp =
                &Kb[(rowBase + nt * 16 + r) * HD + colH + kk * 32 + h * 16];
            bk.q[0] = *(const uint4*)(kp);
            bk.q[1] = *(const uint4*)(kp + 8);
            acc = __builtin_amdgcn_wmma_f32_16x16x32_bf16(
                false, aq[kk].v, false, bk.v, (short)0, acc, false, false);
        }
        int n = nt * 16 + r;
        #pragma unroll
        for (int vg = 0; vg < 8; ++vg) {
            int m = wid * 16 + vg + h * 8;
            size_t mi = ((size_t)bb * NSEQ + q0 + m) * NSEQ + n;
            bool blocked = (slf[mi] != 0);
            if (useAdj) blocked = blocked || (adj[mi] == 0);
            Ssm[m * 264 + n] = blocked ? kNeg : acc[vg] * kScale;
        }
    }
    __syncthreads();

    // row softmax (one thread per query row)
    if (tid < 64) {
        float* srow = &Ssm[tid * 264];
        float mx = -3.4e38f;
        for (int c = 0; c < NSEQ; ++c) mx = fmaxf(mx, srow[c]);
        float sum = 0.f;
        for (int c = 0; c < NSEQ; ++c) { float e = __expf(srow[c] - mx); srow[c] = e; sum += e; }
        float inv = 1.f / sum;
        for (int c = 0; c < NSEQ; ++c) srow[c] *= inv;
    }
    __syncthreads();

    // out = P @ V  (4 dk tiles of 16, K loop over 256 keys)
    f32x8 acc[4];
    #pragma unroll
    for (int tj = 0; tj < 4; ++tj) acc[tj] = zero8();

    for (int kc = 0; kc < NSEQ; kc += 32) {
        FragB16 ap;   // convert f32 probs -> bf16 fragment on the fly
        const float* sp = &Ssm[(wid * 16 + r) * 264 + kc];
        #pragma unroll
        for (int j = 0; j < 8; ++j) ap.s[j]     = f2b(sp[h * 16 + j]);
        #pragma unroll
        for (int j = 0; j < 8; ++j) ap.s[8 + j] = f2b(sp[h * 16 + 8 + j]);
        #pragma unroll
        for (int tj = 0; tj < 4; ++tj) {
            FragB16 bv;   // lane = dk col tj*16+r, contiguous keys at kc+h*16
            const unsigned short* vp = &Vt[(tj * 16 + r) * 264 + kc + h * 16];
            bv.q[0] = *(const uint4*)(vp);
            bv.q[1] = *(const uint4*)(vp + 8);
            acc[tj] = __builtin_amdgcn_wmma_f32_16x16x32_bf16(
                false, ap.v, false, bv.v, (short)0, acc[tj], false, false);
        }
    }

    #pragma unroll
    for (int tj = 0; tj < 4; ++tj)
        #pragma unroll
        for (int vg = 0; vg < 8; ++vg) {
            int m = wid * 16 + vg + h * 8;
            Ctx[(rowBase + q0 + m) * HD + colH + tj * 16 + r] = f2b(acc[tj][vg]);
        }
}

// ---------------- residual + layernorm + non-pad mask ------------------------
// one block (256 threads) per row; writes f32 (opt) and bf16 (opt, with colOff)
__global__ __launch_bounds__(256) void ln_kernel(
    const float* __restrict__ xin, const float* __restrict__ resid,
    const float* __restrict__ gamma, const float* __restrict__ beta,
    const float* __restrict__ npm,
    float* __restrict__ outF,
    unsigned short* __restrict__ outB, int ldB, int colOff)
{
    __shared__ float red[256];
    __shared__ float red2[256];
    const int row = blockIdx.x;
    const int tid = threadIdx.x;
    float v[4], s = 0.f, s2 = 0.f;
    #pragma unroll
    for (int j = 0; j < 4; ++j) {
        int c = tid + j * 256;
        float t = xin[(size_t)row * DMODEL + c] + resid[(size_t)row * DMODEL + c];
        v[j] = t; s += t; s2 += t * t;
    }
    red[tid] = s; red2[tid] = s2; __syncthreads();
    for (int off = 128; off > 0; off >>= 1) {
        if (tid < off) { red[tid] += red[tid + off]; red2[tid] += red2[tid + off]; }
        __syncthreads();
    }
    float mean = red[0] * (1.f / DMODEL);
    float var  = red2[0] * (1.f / DMODEL) - mean * mean;
    float rstd = rsqrtf(var + 1e-5f);
    float np   = npm[row];
    #pragma unroll
    for (int j = 0; j < 4; ++j) {
        int c = tid + j * 256;
        float y = ((v[j] - mean) * rstd * gamma[c] + beta[c]) * np;
        if (outF) outF[(size_t)row * DMODEL + c] = y;
        if (outB) outB[(size_t)row * ldB + colOff + c] = f2b(y);
    }
}

// ============================================================================
extern "C" void kernel_launch(void* const* d_in, const int* in_sizes, int n_in,
                              void* d_out, int out_size, void* d_ws, size_t ws_size,
                              hipStream_t stream) {
    (void)in_sizes; (void)n_in; (void)out_size; (void)ws_size;

    const float* x_f  = (const float*)d_in[0];
    const int*   adj  = (const int*)d_in[1];
    const float* npm  = (const float*)d_in[2];
    const int*   slf  = (const int*)d_in[3];
    const float* fc_w = (const float*)d_in[30];
    const float* fc_b = (const float*)d_in[31];

    char* ws = (char*)d_ws;
    size_t off = 0;
    auto wsAlloc = [&](size_t bytes) -> void* {
        void* p = (void*)(ws + off);
        off += (bytes + 255) & ~(size_t)255;
        return p;
    };

    unsigned short* xb    = (unsigned short*)wsAlloc((size_t)BN * DMODEL * 2);
    unsigned short* wqT   = (unsigned short*)wsAlloc((size_t)HD * DMODEL * 2);
    unsigned short* wkT   = (unsigned short*)wsAlloc((size_t)HD * DMODEL * 2);
    unsigned short* wvT   = (unsigned short*)wsAlloc((size_t)HD * DMODEL * 2);
    unsigned short* woT   = (unsigned short*)wsAlloc((size_t)DMODEL * HD * 2);
    unsigned short* w1T   = (unsigned short*)wsAlloc((size_t)DINNER * DMODEL * 2);
    unsigned short* w2T   = (unsigned short*)wsAlloc((size_t)DMODEL * DINNER * 2);
    unsigned short* fcT   = (unsigned short*)wsAlloc((size_t)DMODEL * 2 * DMODEL * 2);
    unsigned short* qb    = (unsigned short*)wsAlloc((size_t)BN * HD * 2);
    unsigned short* kb    = (unsigned short*)wsAlloc((size_t)BN * HD * 2);
    unsigned short* vb    = (unsigned short*)wsAlloc((size_t)BN * HD * 2);
    unsigned short* ctx   = (unsigned short*)wsAlloc((size_t)BN * HD * 2);
    float*          t0    = (float*)wsAlloc((size_t)BN * DMODEL * 4);
    float*          out1  = (float*)wsAlloc((size_t)BN * DMODEL * 4);
    unsigned short* out1b = (unsigned short*)wsAlloc((size_t)BN * DMODEL * 2);
    unsigned short* h1    = (unsigned short*)wsAlloc((size_t)BN * DINNER * 2);
    unsigned short* cat   = (unsigned short*)wsAlloc((size_t)BN * 2 * DMODEL * 2);

    // x -> bf16 once
    {
        int n = BN * DMODEL;
        cvt_kernel<<<(n + 255) / 256, 256, 0, stream>>>(x_f, xb, n);
    }
    // fc_w^T once
    {
        int n = 2 * DMODEL * DMODEL;
        tcvt_kernel<<<(n + 255) / 256, 256, 0, stream>>>(fc_w, fcT, 2 * DMODEL, DMODEL);
    }

    for (int t = 0; t < 2; ++t) {
        const int base = 4 + 13 * t;
        const float* wq   = (const float*)d_in[base + 0];
        const float* wk   = (const float*)d_in[base + 1];
        const float* wv   = (const float*)d_in[base + 2];
        const float* wo   = (const float*)d_in[base + 3];
        const float* bo   = (const float*)d_in[base + 4];
        const float* ln1s = (const float*)d_in[base + 5];
        const float* ln1b = (const float*)d_in[base + 6];
        const float* w1   = (const float*)d_in[base + 7];
        const float* b1   = (const float*)d_in[base + 8];
        const float* w2   = (const float*)d_in[base + 9];
        const float* b2   = (const float*)d_in[base + 10];
        const float* ln2s = (const float*)d_in[base + 11];
        const float* ln2b = (const float*)d_in[base + 12];

        // weight transposes -> bf16
        { int n = DMODEL * HD;     tcvt_kernel<<<(n + 255) / 256, 256, 0, stream>>>(wq, wqT, DMODEL, HD); }
        { int n = DMODEL * HD;     tcvt_kernel<<<(n + 255) / 256, 256, 0, stream>>>(wk, wkT, DMODEL, HD); }
        { int n = DMODEL * HD;     tcvt_kernel<<<(n + 255) / 256, 256, 0, stream>>>(wv, wvT, DMODEL, HD); }
        { int n = HD * DMODEL;     tcvt_kernel<<<(n + 255) / 256, 256, 0, stream>>>(wo, woT, HD, DMODEL); }
        { int n = DMODEL * DINNER; tcvt_kernel<<<(n + 255) / 256, 256, 0, stream>>>(w1, w1T, DMODEL, DINNER); }
        { int n = DINNER * DMODEL; tcvt_kernel<<<(n + 255) / 256, 256, 0, stream>>>(w2, w2T, DINNER, DMODEL); }

        // QKV projections: [BN,1024] x [1024,512]
        dim3 gQKV(HD / 128, BN / 128);
        gemm_bf16_kernel<<<gQKV, 256, 0, stream>>>(xb, DMODEL, wqT, DMODEL, nullptr,
                                                   nullptr, qb, HD, 0, DMODEL, 0);
        gemm_bf16_kernel<<<gQKV, 256, 0, stream>>>(xb, DMODEL, wkT, DMODEL, nullptr,
                                                   nullptr, kb, HD, 0, DMODEL, 0);
        gemm_bf16_kernel<<<gQKV, 256, 0, stream>>>(xb, DMODEL, wvT, DMODEL, nullptr,
                                                   nullptr, vb, HD, 0, DMODEL, 0);

        // fused masked attention
        attn_kernel<<<BDIM * NHEAD * 4, 128, 0, stream>>>(qb, kb, vb, slf, adj,
                                                          t == 1 ? 1 : 0, ctx);

        // output projection: [BN,512] x [512,1024] + bo -> t0 (f32)
        dim3 gWo(DMODEL / 128, BN / 128);
        gemm_bf16_kernel<<<gWo, 256, 0, stream>>>(ctx, HD, woT, HD, bo,
                                                  t0, nullptr, DMODEL, 0, HD, 0);

        // out1 = LN(t0 + x) * npm -> f32 + bf16
        ln_kernel<<<BN, 256, 0, stream>>>(t0, x_f, ln1s, ln1b, npm,
                                          out1, out1b, DMODEL, 0);

        // h1 = relu(out1 @ w1 + b1) -> bf16 [BN,2048]
        dim3 gW1(DINNER / 128, BN / 128);
        gemm_bf16_kernel<<<gW1, 256, 0, stream>>>(out1b, DMODEL, w1T, DMODEL, b1,
                                                  nullptr, h1, DINNER, 0, DMODEL, 1);

        // t0 = h1 @ w2 + b2 -> f32 [BN,1024]
        dim3 gW2(DMODEL / 128, BN / 128);
        gemm_bf16_kernel<<<gW2, 256, 0, stream>>>(h1, DINNER, w2T, DINNER, b2,
                                                  t0, nullptr, DMODEL, 0, DINNER, 0);

        // x_t = LN(t0 + out1) * npm -> bf16 into concat[:, t*1024:]
        ln_kernel<<<BN, 256, 0, stream>>>(t0, out1, ln2s, ln2b, npm,
                                          nullptr, cat, 2 * DMODEL, t * DMODEL);
    }

    // final: out = concat @ fc_w + fc_b -> f32 d_out [BN,1024]
    dim3 gFc(DMODEL / 128, BN / 128);
    gemm_bf16_kernel<<<gFc, 256, 0, stream>>>(cat, 2 * DMODEL, fcT, 2 * DMODEL, fc_b,
                                              (float*)d_out, nullptr, DMODEL, 0,
                                              2 * DMODEL, 0);
}
